// TradingTransformer_7584912244961
// MI455X (gfx1250) — matrix operand order repaired and checked by reading, not compiled, not run on hardware
//
#include <hip/hip_runtime.h>

// ---------------------------------------------------------------------------
// Transformer encoder forward for gfx1250 (MI455X), WMMA f16 x f16 -> f32.
// Tile staging uses CDNA5 async global->LDS DMA (ASYNCcnt) + prefetch.
// Workspace requirement: ~216 MB (see layout below).
// ---------------------------------------------------------------------------

typedef _Float16 half_t;
typedef __attribute__((ext_vector_type(16))) _Float16 v16h;
typedef __attribute__((ext_vector_type(8)))  float    v8f;

static constexpr int B_   = 16;
static constexpr int S_   = 1024;
static constexpr int IN_  = 64;
static constexpr int D_   = 512;
static constexpr int H_   = 8;
static constexpr int L_   = 6;
static constexpr int DFF_ = 2048;
static constexpr int DK_  = 64;
static constexpr int BH_  = B_ * H_;        // 128
static constexpr int MTOT_= B_ * S_;        // 16384
static constexpr int CH_  = 8;              // (b,h) pairs per attention chunk

// Async copy of 16 bytes per lane: global -> LDS, tracked by ASYNCcnt.
// lds_addr is the raw LDS byte offset (low 32 bits of the generic pointer).
__device__ __forceinline__ void async_copy_b128(unsigned lds_addr, const void* gptr) {
    unsigned long long ga = (unsigned long long)(uintptr_t)gptr;
    asm volatile("global_load_async_to_lds_b128 %0, %1, off"
                 :: "v"(lds_addr), "v"(ga) : "memory");
}
__device__ __forceinline__ void wait_async0() {
    asm volatile("s_wait_asynccnt 0x0" ::: "memory");
}

// ---------------------------------------------------------------------------
// Generic tiled WMMA GEMM:  C = act( alpha * A[M,K] @ Bt[N,K]^T + bias )
//   A  : f16 row-major [M,K]  (per-batch stride aBatch)
//   Bt : f16 row-major [N,K]  (i.e. B transposed; per-batch stride bBatch)
//   C  : optional f32 row-major [M,ldc]
//   C16: optional f16 output with layout mode:
//        0: row-major [M,ld16]
//        1: head-split   dst = ((b*H+h)*S+s)*DK+d   (m=b*S+s, n=h*DK+d)
//        2: V-transpose  dst = ((b*H+h)*DK+d)*S+s
//        3: attn merge   dst = (b*S+m)*D + h*DK+n   (bh = bh0 + blockIdx.z)
// Block: 256 threads (8 waves). Block tile 128x128, K step 32.
// Wave grid 4(M) x 2(N): each wave computes 32x64 = 2x4 WMMA tiles.
// Requires K % 32 == 0 (true for all uses: 64/512/1024/2048).
// NOTE: tile loads are unguarded. Out-of-range tile rows occur only for
// Bt in the attn@V GEMM (N=64 < 128); those reads remain inside the
// workspace, and garbage columns/rows only feed output elements that the
// guarded epilogue never stores.
// ---------------------------------------------------------------------------
__global__ __launch_bounds__(256) void gemm_wmma_f16(
    const half_t* __restrict__ A, const half_t* __restrict__ Bt,
    const float* __restrict__ bias, float* __restrict__ C,
    half_t* __restrict__ C16,
    int M, int N, int K, int lda, int ldb, int ldc, int ld16,
    long long aBatch, long long bBatch, long long cBatch,
    float alpha, int relu, int c16mode, int bh0)
{
    __shared__ half_t As[128][40];   // pad to 40 halves: 80B row stride (16B aligned)
    __shared__ half_t Bs[128][40];   // Bs[n][k]

    const int z = blockIdx.z;
    A  += (long long)z * aBatch;
    Bt += (long long)z * bBatch;
    if (C) C += (long long)z * cBatch;

    const int bm   = blockIdx.y * 128;
    const int bn   = blockIdx.x * 128;
    const int tid  = threadIdx.x;
    const int lane = tid & 31;
    const int wave = tid >> 5;
    const int mw   = (wave & 3) * 32;   // wave M origin in block tile
    const int nw   = (wave >> 2) * 64;  // wave N origin in block tile
    const int hlf  = lane >> 4;         // 0/1 : lane half
    const int l15  = lane & 15;

    v8f acc[2][4];
#pragma unroll
    for (int i = 0; i < 2; ++i)
#pragma unroll
        for (int j = 0; j < 4; ++j)
#pragma unroll
            for (int e = 0; e < 8; ++e) acc[i][j][e] = 0.0f;

    const int lrow = tid >> 2;         // 0..63
    const int lcol = (tid & 3) * 8;    // 0/8/16/24

    // per-thread source rows (A rows bm+lrow, bm+lrow+64; Bt rows bn+...)
    const half_t* aRow0 = A + (long long)(bm + lrow) * lda + lcol;
    const half_t* aRow1 = A + (long long)(bm + lrow + 64) * lda + lcol;
    const half_t* bRow0 = Bt + (long long)(bn + lrow) * ldb + lcol;
    const half_t* bRow1 = Bt + (long long)(bn + lrow + 64) * ldb + lcol;
    const unsigned ldsA0 = (unsigned)(uintptr_t)&As[lrow][lcol];
    const unsigned ldsA1 = (unsigned)(uintptr_t)&As[lrow + 64][lcol];
    const unsigned ldsB0 = (unsigned)(uintptr_t)&Bs[lrow][lcol];
    const unsigned ldsB1 = (unsigned)(uintptr_t)&Bs[lrow + 64][lcol];

    for (int k0 = 0; k0 < K; k0 += 32) {
        // ---- stage A/B tiles via async global->LDS DMA ----
        async_copy_b128(ldsA0, aRow0 + k0);
        async_copy_b128(ldsA1, aRow1 + k0);
        async_copy_b128(ldsB0, bRow0 + k0);
        async_copy_b128(ldsB1, bRow1 + k0);
        if (k0 + 32 < K) {      // warm L2 for the next K tile (global_prefetch_b8)
            __builtin_prefetch(aRow0 + k0 + 32, 0, 1);
            __builtin_prefetch(bRow0 + k0 + 32, 0, 1);
        }
        wait_async0();
        __syncthreads();

        // ---- build fragments per ISA 16-bit layouts (wave32) ----
        v16h afrag[2], bfrag[4];
#pragma unroll
        for (int mt = 0; mt < 2; ++mt) {
            int row = mw + mt * 16 + l15;        // M = lane%16
            union { v16h v; unsigned u[8]; } r;
#pragma unroll
            for (int j = 0; j < 8; ++j) {
                // A 16x32: vgpr j -> K = (j/4)*16 + half*8 + (j%4)*2  (pair)
                int k = ((j >> 2) << 4) + (hlf << 3) + ((j & 3) << 1);
                r.u[j] = *reinterpret_cast<const unsigned*>(&As[row][k]);
            }
            afrag[mt] = r.v;
        }
#pragma unroll
        for (int nt = 0; nt < 4; ++nt) {
            int col = nw + nt * 16 + l15;        // N = lane%16
            union { v16h v; unsigned u[8]; } r;
#pragma unroll
            for (int j = 0; j < 8; ++j) {
                // B 32x16: vgpr j -> K = (j/4)*16 + half*8 + (j%4)*2  (pair), same as A
                int k = ((j >> 2) << 4) + (hlf << 3) + ((j & 3) << 1);
                r.u[j] = *reinterpret_cast<const unsigned*>(&Bs[col][k]);
            }
            bfrag[nt] = r.v;
        }

#pragma unroll
        for (int mt = 0; mt < 2; ++mt)
#pragma unroll
            for (int nt = 0; nt < 4; ++nt)
                acc[mt][nt] = __builtin_amdgcn_wmma_f32_16x16x32_f16(
                    false, afrag[mt], false, bfrag[nt],
                    (short)0, acc[mt][nt], false, false);
        __syncthreads();
    }

    // ---- epilogue: C/D layout -> VGPR i holds M=i (lanes 0-15) / M=8+i (16-31)
#pragma unroll
    for (int mt = 0; mt < 2; ++mt) {
#pragma unroll
        for (int nt = 0; nt < 4; ++nt) {
#pragma unroll
            for (int i = 0; i < 8; ++i) {
                int m = bm + mw + mt * 16 + hlf * 8 + i;
                int n = bn + nw + nt * 16 + l15;
                if (m < M && n < N) {
                    float v = acc[mt][nt][i] * alpha;
                    if (bias) v += bias[n];
                    if (relu) v = fmaxf(v, 0.0f);
                    if (C) C[(long long)m * ldc + n] = v;
                    if (C16) {
                        long long dst;
                        if (c16mode == 0) {
                            dst = (long long)m * ld16 + n;
                        } else if (c16mode == 1) {
                            int b = m / S_, s = m % S_, hh = n / DK_, d = n % DK_;
                            dst = (((long long)(b * H_ + hh)) * S_ + s) * DK_ + d;
                        } else if (c16mode == 2) {
                            int b = m / S_, s = m % S_, hh = n / DK_, d = n % DK_;
                            dst = (((long long)(b * H_ + hh)) * DK_ + d) * S_ + s;
                        } else {
                            int bh = bh0 + z; int b = bh / H_, hh = bh % H_;
                            dst = ((long long)(b * S_ + m)) * D_ + hh * DK_ + n;
                        }
                        C16[dst] = (half_t)v;
                    }
                }
            }
        }
    }
}

// ---------------------------------------------------------------------------
// Small helper kernels
// ---------------------------------------------------------------------------
__global__ void f32_to_f16_kernel(const float* __restrict__ src,
                                  half_t* __restrict__ dst, long long n) {
    long long i = (long long)blockIdx.x * 256 + threadIdx.x;
    if (i < n) dst[i] = (half_t)src[i];
}

// w[K][N] -> wt[N][K] as f16
__global__ void transpose_w_kernel(const float* __restrict__ w,
                                   half_t* __restrict__ wt, int K, int N) {
    long long i = (long long)blockIdx.x * 256 + threadIdx.x;
    if (i >= (long long)K * N) return;
    int k = (int)(i / N), n = (int)(i % N);
    wt[(long long)n * K + k] = (half_t)w[i];
}

// h[(b*S+s)*D+d] += pe[s*D+d]; also emit f16 copy
__global__ void add_pe_kernel(float* __restrict__ h, const float* __restrict__ pe,
                              half_t* __restrict__ h16) {
    long long i = (long long)blockIdx.x * 256 + threadIdx.x;
    if (i >= (long long)MTOT_ * D_) return;
    int d = (int)(i % D_);
    int s = (int)((i / D_) % S_);
    float v = h[i] + pe[s * D_ + d];
    h[i] = v;
    h16[i] = (half_t)v;
}

// out = LN(x + y) * g + be ; one block (256 thr) per row of D=512; emits f32+f16
__global__ __launch_bounds__(256) void ln_residual_kernel(
    const float* __restrict__ x, const float* __restrict__ y,
    const float* __restrict__ g, const float* __restrict__ be,
    float* __restrict__ out, half_t* __restrict__ out16) {
    long long row = blockIdx.x;
    const float* xr = x + row * D_;
    const float* yr = y + row * D_;
    int t = threadIdx.x;
    float v0 = xr[t] + yr[t];
    float v1 = xr[t + 256] + yr[t + 256];
    float s = v0 + v1, ss = v0 * v0 + v1 * v1;
    for (int off = 16; off; off >>= 1) {
        s  += __shfl_xor(s,  off, 32);
        ss += __shfl_xor(ss, off, 32);
    }
    __shared__ float rs[8], rss[8];
    int wave = t >> 5, lane = t & 31;
    if (lane == 0) { rs[wave] = s; rss[wave] = ss; }
    __syncthreads();
    if (wave == 0) {
        s  = (lane < 8) ? rs[lane]  : 0.0f;
        ss = (lane < 8) ? rss[lane] : 0.0f;
        for (int off = 4; off; off >>= 1) {
            s  += __shfl_xor(s,  off, 32);
            ss += __shfl_xor(ss, off, 32);
        }
        if (lane == 0) { rs[0] = s; rss[0] = ss; }
    }
    __syncthreads();
    float mu  = rs[0] * (1.0f / D_);
    float var = rss[0] * (1.0f / D_) - mu * mu;
    float r   = rsqrtf(var + 1e-5f);
    float o0 = (v0 - mu) * r * g[t] + be[t];
    float o1 = (v1 - mu) * r * g[t + 256] + be[t + 256];
    out[row * D_ + t] = o0;          out16[row * D_ + t] = (half_t)o0;
    out[row * D_ + t + 256] = o1;    out16[row * D_ + t + 256] = (half_t)o1;
}

// rowwise softmax over 1024 f32 -> f16 probs; one block per row
__global__ __launch_bounds__(256) void softmax_kernel(
    const float* __restrict__ s, half_t* __restrict__ p) {
    long long row = blockIdx.x;
    const float* sr = s + row * S_;
    half_t* pr = p + row * S_;
    int t = threadIdx.x;
    float v[4]; float mx = -1e30f;
#pragma unroll
    for (int i = 0; i < 4; ++i) { v[i] = sr[t + i * 256]; mx = fmaxf(mx, v[i]); }
    for (int off = 16; off; off >>= 1) mx = fmaxf(mx, __shfl_xor(mx, off, 32));
    __shared__ float red[8];
    int wave = t >> 5, lane = t & 31;
    if (lane == 0) red[wave] = mx;
    __syncthreads();
    if (wave == 0) {
        mx = (lane < 8) ? red[lane] : -1e30f;
        for (int off = 4; off; off >>= 1) mx = fmaxf(mx, __shfl_xor(mx, off, 32));
        if (lane == 0) red[0] = mx;
    }
    __syncthreads();
    mx = red[0];
    float sum = 0.0f;
#pragma unroll
    for (int i = 0; i < 4; ++i) { v[i] = expf(v[i] - mx); sum += v[i]; }
    for (int off = 16; off; off >>= 1) sum += __shfl_xor(sum, off, 32);
    __syncthreads();
    if (lane == 0) red[wave] = sum;
    __syncthreads();
    if (wave == 0) {
        sum = (lane < 8) ? red[lane] : 0.0f;
        for (int off = 4; off; off >>= 1) sum += __shfl_xor(sum, off, 32);
        if (lane == 0) red[0] = sum;
    }
    __syncthreads();
    float inv = 1.0f / red[0];
#pragma unroll
    for (int i = 0; i < 4; ++i) pr[t + i * 256] = (half_t)(v[i] * inv);
}

// pooled[b][0:D)=mean, [D:2D)=max, [2D:3D)=sum over S of h
__global__ void pool_kernel(const float* __restrict__ h, float* __restrict__ pooled) {
    int idx = blockIdx.x * 256 + threadIdx.x;    // b*D + d
    if (idx >= B_ * D_) return;
    int b = idx / D_, d = idx % D_;
    const float* p = h + ((long long)b * S_) * D_ + d;
    float sum = 0.0f, mx = -1e30f;
    for (int s = 0; s < S_; ++s) {
        float v = p[(long long)s * D_];
        sum += v; mx = fmaxf(mx, v);
    }
    pooled[b * 3 * D_ + d]         = sum * (1.0f / S_);
    pooled[b * 3 * D_ + D_ + d]    = mx;
    pooled[b * 3 * D_ + 2 * D_ + d]= sum;
}

// tiny FC (classifier head): out[m][n] = act(A[m,:] @ W[:,n] + b[n])
__global__ void fc_kernel(const float* __restrict__ A, const float* __restrict__ W,
                          const float* __restrict__ b, float* __restrict__ out,
                          int M, int K, int N, int relu) {
    int idx = blockIdx.x * 256 + threadIdx.x;
    if (idx >= M * N) return;
    int m = idx / N, n = idx % N;
    float s = b ? b[n] : 0.0f;
    for (int k = 0; k < K; ++k) s += A[(long long)m * K + k] * W[(long long)k * N + n];
    if (relu) s = fmaxf(s, 0.0f);
    out[idx] = s;
}

// ---------------------------------------------------------------------------
// Host orchestration
// ---------------------------------------------------------------------------
static void run_gemm(hipStream_t st,
                     const half_t* A, const half_t* Bt, const float* bias,
                     float* C, half_t* C16,
                     int M, int N, int K, int lda, int ldb, int ldc, int ld16,
                     long long aB, long long bB, long long cB,
                     float alpha, int relu, int mode, int bh0, int batches) {
    dim3 grid((N + 127) / 128, (M + 127) / 128, batches);
    gemm_wmma_f16<<<grid, 256, 0, st>>>(A, Bt, bias, C, C16, M, N, K,
                                        lda, ldb, ldc, ld16, aB, bB, cB,
                                        alpha, relu, mode, bh0);
}

extern "C" void kernel_launch(void* const* d_in, const int* in_sizes, int n_in,
                              void* d_out, int out_size, void* d_ws, size_t ws_size,
                              hipStream_t stream) {
    (void)in_sizes; (void)n_in; (void)out_size;
    const float* x    = (const float*)d_in[0];
    const float* w_in = (const float*)d_in[1];
    const float* b_in = (const float*)d_in[2];
    const float* pe   = (const float*)d_in[3];
    auto LW = [&](int l, int j) { return (const float*)d_in[4 + l * 16 + j]; };
    const float* w_pool = (const float*)d_in[100];
    const float* b_pool = (const float*)d_in[101];
    const float* w_c1   = (const float*)d_in[102];
    const float* b_c1   = (const float*)d_in[103];
    const float* w_c2   = (const float*)d_in[104];
    const float* b_c2   = (const float*)d_in[105];

    // ---- workspace layout (bytes); total ~216 MB ----
    char* base = (char*)d_ws;
    const long long MB = 1ll << 20;
    float*  h      = (float*) (base + 0 * MB);      // 32 MB  [MTOT, D] f32
    half_t* h16    = (half_t*)(base + 32 * MB);     // 16 MB  f16 copy of h
    float*  t0     = (float*) (base + 48 * MB);     // 32 MB  projection scratch f32
    half_t* q16    = (half_t*)(base + 80 * MB);     // 16 MB  [BH,S,DK]
    half_t* k16    = (half_t*)(base + 96 * MB);     // 16 MB  [BH,S,DK]
    half_t* vt16   = (half_t*)(base + 112 * MB);    // 16 MB  [BH,DK,S]
    half_t* attn16 = (half_t*)(base + 128 * MB);    // 16 MB  merged attn out f16 [MTOT,D]
    half_t* ff16   = (half_t*)(base + 144 * MB);    // 64 MB  [MTOT,DFF] f16
    // scores / probs overlap the (dead-at-that-point) ff16 region:
    float*  scores = (float*) (base + 144 * MB);    // 32 MB  [CH,S,S] f32
    half_t* p16    = (half_t*)(base + 176 * MB);    // 16 MB  [CH,S,S] f16
    half_t* wbuf   = (half_t*)(base + 208 * MB);    // 4 MB   transposed f16 weights
    float*  pooled = (float*) (base + 212 * MB);    //        [B, 3D]
    float*  pout   = (float*) (base + 213 * MB);    //        [B, D]
    float*  c1out  = (float*) (base + 213 * MB + 64 * 1024); // [B, 256]
    half_t* x16    = attn16;                        // reuse for input conversion
    (void)ws_size;

    const long long nSD = (long long)S_ * DK_;      // per-(b,h) q/k stride
    const long long nSS = (long long)S_ * S_;       // per-(b,h) score stride

    // ---- input projection + positional encoding ----
    {
        long long nx = (long long)MTOT_ * IN_;
        f32_to_f16_kernel<<<(unsigned)((nx + 255) / 256), 256, 0, stream>>>(x, x16, nx);
        long long nw = (long long)IN_ * D_;
        transpose_w_kernel<<<(unsigned)((nw + 255) / 256), 256, 0, stream>>>(w_in, wbuf, IN_, D_);
        run_gemm(stream, x16, wbuf, b_in, h, nullptr,
                 MTOT_, D_, IN_, IN_, IN_, D_, 0, 0, 0, 0, 1.0f, 0, 0, 0, 1);
        long long nh = (long long)MTOT_ * D_;
        add_pe_kernel<<<(unsigned)((nh + 255) / 256), 256, 0, stream>>>(h, pe, h16);
    }

    // ---- transformer layers ----
    for (int l = 0; l < L_; ++l) {
        const float *wq = LW(l, 0), *bq = LW(l, 1), *wk = LW(l, 2), *bk = LW(l, 3);
        const float *wv = LW(l, 4), *bv = LW(l, 5), *wo = LW(l, 6), *bo = LW(l, 7);
        const float *g1 = LW(l, 8), *be1 = LW(l, 9), *g2 = LW(l, 10), *be2 = LW(l, 11);
        const float *w1 = LW(l, 12), *bf1 = LW(l, 13), *w2 = LW(l, 14), *bf2 = LW(l, 15);
        long long nDD = (long long)D_ * D_;
        unsigned gDD = (unsigned)((nDD + 255) / 256);

        // Q, K (head-split f16), V (transposed f16) projections
        transpose_w_kernel<<<gDD, 256, 0, stream>>>(wq, wbuf, D_, D_);
        run_gemm(stream, h16, wbuf, bq, nullptr, q16,
                 MTOT_, D_, D_, D_, D_, 0, 0, 0, 0, 0, 1.0f, 0, 1, 0, 1);
        transpose_w_kernel<<<gDD, 256, 0, stream>>>(wk, wbuf, D_, D_);
        run_gemm(stream, h16, wbuf, bk, nullptr, k16,
                 MTOT_, D_, D_, D_, D_, 0, 0, 0, 0, 0, 1.0f, 0, 1, 0, 1);
        transpose_w_kernel<<<gDD, 256, 0, stream>>>(wv, wbuf, D_, D_);
        run_gemm(stream, h16, wbuf, bv, nullptr, vt16,
                 MTOT_, D_, D_, D_, D_, 0, 0, 0, 0, 0, 1.0f, 0, 2, 0, 1);

        // attention, CH_ (b,h) pairs at a time
        for (int c = 0; c < BH_ / CH_; ++c) {
            int bh0 = c * CH_;
            // scores = (1/sqrt(DK)) * Q @ K^T   (K^T supplied as Bt=[N=S][K=DK])
            run_gemm(stream, q16 + (long long)bh0 * nSD, k16 + (long long)bh0 * nSD,
                     nullptr, scores, nullptr,
                     S_, S_, DK_, DK_, DK_, S_, 0, nSD, nSD, nSS,
                     0.125f, 0, 0, 0, CH_);
            softmax_kernel<<<CH_ * S_, 256, 0, stream>>>(scores, p16);
            // out = P @ V   (V^T supplied as Bt=[N=DK][K=S]); merge into attn16
            run_gemm(stream, p16, vt16 + (long long)bh0 * nSD, nullptr,
                     nullptr, attn16,
                     S_, DK_, S_, S_, S_, 0, 0, nSS, nSD, 0,
                     1.0f, 0, 3, bh0, CH_);
        }

        // output projection + residual LN1
        transpose_w_kernel<<<gDD, 256, 0, stream>>>(wo, wbuf, D_, D_);
        run_gemm(stream, attn16, wbuf, bo, t0, nullptr,
                 MTOT_, D_, D_, D_, D_, D_, 0, 0, 0, 0, 1.0f, 0, 0, 0, 1);
        ln_residual_kernel<<<MTOT_, 256, 0, stream>>>(h, t0, g1, be1, h, h16);

        // FFN
        long long nDF = (long long)D_ * DFF_;
        unsigned gDF = (unsigned)((nDF + 255) / 256);
        transpose_w_kernel<<<gDF, 256, 0, stream>>>(w1, wbuf, D_, DFF_);
        run_gemm(stream, h16, wbuf, bf1, nullptr, ff16,
                 MTOT_, DFF_, D_, D_, D_, 0, DFF_, 0, 0, 0, 1.0f, 1, 0, 0, 1);
        transpose_w_kernel<<<gDF, 256, 0, stream>>>(w2, wbuf, DFF_, D_);
        run_gemm(stream, ff16, wbuf, bf2, t0, nullptr,
                 MTOT_, D_, DFF_, DFF_, DFF_, D_, 0, 0, 0, 0, 1.0f, 0, 0, 0, 1);
        ln_residual_kernel<<<MTOT_, 256, 0, stream>>>(h, t0, g2, be2, h, h16);
    }

    // ---- pooling + classifier head ----
    pool_kernel<<<(B_ * D_ + 255) / 256, 256, 0, stream>>>(h, pooled);
    fc_kernel<<<(B_ * D_ + 255) / 256, 256, 0, stream>>>(
        pooled, w_pool, b_pool, pout, B_, 3 * D_, D_, 1);
    fc_kernel<<<(B_ * 256 + 255) / 256, 256, 0, stream>>>(
        pout, w_c1, b_c1, c1out, B_, D_, 256, 1);
    fc_kernel<<<1, 256, 0, stream>>>(
        c1out, w_c2, b_c2, (float*)d_out, B_, 256, 1, 0);
}
